// EncoderBlock_13400297964100
// MI455X (gfx1250) — matrix operand order repaired and checked by reading, not compiled, not run on hardware
//
#include <hip/hip_runtime.h>
#include <math.h>

#define DEVFN __device__ __forceinline__

namespace {
constexpr int Nn    = 32;
constexpr int Gg    = 197;
constexpr int Cc    = 768;
constexpr int HEADS = 12;
constexpr int Dd    = 64;
constexpr int MLPd  = 3072;
constexpr int Mrows = Nn * Gg;          // 6304 = 394 * 16
constexpr int HHc   = 14;
constexpr int Fc    = 8;
constexpr int MIXO  = HHc * Fc * Cc;    // 86016 complex outputs of mixer reduction
}

typedef __attribute__((ext_vector_type(16))) _Float16 v16h;
typedef __attribute__((ext_vector_type(8)))  _Float16 v8h;
typedef __attribute__((ext_vector_type(8)))  float    v8f;

// ---------------------------------------------------------------------------
// K1: m_eff[h,f,c] = sum_s mixer[h,f,c,s,:]   (528 MB read -> 688 KB)
// One wave per (h,f,c); lanes stride the 768 complex values (float2 loads).
// ---------------------------------------------------------------------------
__global__ __launch_bounds__(256) void mixsum_kernel(const float* __restrict__ mixer,
                                                     float* __restrict__ meff) {
  const int o    = blockIdx.x * 8 + (threadIdx.x >> 5);
  const int lane = threadIdx.x & 31;
  if (o >= MIXO) return;
  const float2* p = (const float2*)(mixer + (size_t)o * (size_t)(2 * Cc));
  float re = 0.f, im = 0.f;
  for (int i = lane; i < Cc; i += 32) {
    float2 v = p[i];
    re += v.x; im += v.y;
  }
  for (int off = 16; off > 0; off >>= 1) {
    re += __shfl_down(re, off, 32);
    im += __shfl_down(im, off, 32);
  }
  if (lane == 0) { meff[2 * o] = re; meff[2 * o + 1] = im; }
}

// ---------------------------------------------------------------------------
// K2: weight transpose + fp32 -> f16 (RNE via cast). src: K x N, dst: N x K.
// ---------------------------------------------------------------------------
__global__ __launch_bounds__(256) void transpose_f16_kernel(const float* __restrict__ src,
                                                            _Float16* __restrict__ dst,
                                                            int K, int N) {
  __shared__ float tile[32][33];
  const int k0 = blockIdx.x * 32, n0 = blockIdx.y * 32;
  const int tx = threadIdx.x, ty = threadIdx.y;        // blockDim = (32, 8)
  for (int r = ty; r < 32; r += 8)
    tile[r][tx] = src[(size_t)(k0 + r) * N + n0 + tx];
  __syncthreads();
  for (int r = ty; r < 32; r += 8)
    dst[(size_t)(n0 + r) * K + k0 + tx] = (_Float16)tile[tx][r];
}

// ---------------------------------------------------------------------------
// K3: spectral token mixing. One thread = one (n, c) channel.
//   rfft along w (8 bins), full DFT along h, * m_eff, inverse DFT along h,
//   C2R along w (imag of DC/Nyquist dropped, like numpy/XLA irfft).
//   Ortho norms folded into a single 1/196.
// Also copies the cls row (row 0) of x into `full`.
// ---------------------------------------------------------------------------
__global__ __launch_bounds__(64) void spectral_kernel(const float* __restrict__ x,
                                                      const float* __restrict__ meff,
                                                      float* __restrict__ full) {
  const int n = blockIdx.x;
  const int c = blockIdx.y * 64 + threadIdx.x;
  const size_t rowb = (size_t)n * Gg * Cc;

  full[rowb + c] = x[rowb + c];                       // cls passthrough

  float cst[14], snt[14];
#pragma unroll
  for (int m = 0; m < 14; ++m) {
    float a = 6.2831853071795864769f * (float)m / 14.f;
    sincosf(a, &snt[m], &cst[m]);
  }

  float Ur[14][8], Ui[14][8];
#pragma unroll
  for (int h = 0; h < 14; ++h) {
    float row[14];
#pragma unroll
    for (int w = 0; w < 14; ++w)
      row[w] = x[rowb + (size_t)(1 + h * 14 + w) * Cc + c];
#pragma unroll
    for (int kw = 0; kw < 8; ++kw) {
      float ar = 0.f, ai = 0.f;
#pragma unroll
      for (int w = 0; w < 14; ++w) {
        const int m = (w * kw) % 14;
        ar = fmaf(row[w], cst[m], ar);
        ai = fmaf(-row[w], snt[m], ai);
      }
      Ur[h][kw] = ar; Ui[h][kw] = ai;
    }
  }

  float Vr[14][8], Vi[14][8];
#pragma unroll
  for (int h = 0; h < 14; ++h)
#pragma unroll
    for (int kw = 0; kw < 8; ++kw) { Vr[h][kw] = 0.f; Vi[h][kw] = 0.f; }

#pragma unroll
  for (int kh = 0; kh < 14; ++kh) {
#pragma unroll
    for (int kw = 0; kw < 8; ++kw) {
      float xr = 0.f, xi = 0.f;
#pragma unroll
      for (int h = 0; h < 14; ++h) {
        const int m = (h * kh) % 14;               // e^{-i 2pi h kh / 14}
        xr = fmaf(Ur[h][kw], cst[m], fmaf(Ui[h][kw], snt[m], xr));
        xi = fmaf(Ui[h][kw], cst[m], fmaf(-Ur[h][kw], snt[m], xi));
      }
      const size_t mo = ((size_t)(kh * Fc + kw) * Cc + c) * 2;
      const float mr = meff[mo], mi = meff[mo + 1];
      const float yr = xr * mr - xi * mi;
      const float yi = xr * mi + xi * mr;
#pragma unroll
      for (int h = 0; h < 14; ++h) {
        const int m = (h * kh) % 14;               // e^{+i 2pi h kh / 14}
        Vr[h][kw] = fmaf(yr, cst[m], fmaf(-yi, snt[m], Vr[h][kw]));
        Vi[h][kw] = fmaf(yr, snt[m], fmaf(yi, cst[m], Vi[h][kw]));
      }
    }
  }

  const float scale = 1.f / 196.f;
#pragma unroll
  for (int h = 0; h < 14; ++h) {
#pragma unroll
    for (int w = 0; w < 14; ++w) {
      float acc = Vr[h][0] + ((w & 1) ? -Vr[h][7] : Vr[h][7]);
#pragma unroll
      for (int kw = 1; kw < 7; ++kw) {
        const int m = (w * kw) % 14;
        acc = fmaf(2.f * Vr[h][kw], cst[m], fmaf(-2.f * Vi[h][kw], snt[m], acc));
      }
      full[rowb + (size_t)(1 + h * 14 + w) * Cc + c] = acc * scale;
    }
  }
}

// ---------------------------------------------------------------------------
// K4: single-query attention. One block per (n, head). Writes new_cls into
// full[n,0,h*64..] in place (each head touches only its own slice).
//   s[k]   = (full[k]·(Kw^T qhat) + Kb[k]*sum(qhat)) / 8
//   out[x] = inv_sum * ( (sum_k e_k full[k])·Vw[x] + sum_k e_k Vb[k] )
// ---------------------------------------------------------------------------
__global__ __launch_bounds__(256) void attn_kernel(const float* __restrict__ xin,
                                                   const float* __restrict__ Qw,
                                                   const float* __restrict__ Qb,
                                                   const float* __restrict__ Kw,
                                                   const float* __restrict__ Kb,
                                                   const float* __restrict__ Vw,
                                                   const float* __restrict__ Vb,
                                                   float* __restrict__ full) {
  const int n = blockIdx.x, h = blockIdx.y;
  const int tid = threadIdx.x;
  __shared__ float qhat[Dd], wtil[Dd], sc[256], red[256];
  __shared__ float scal[2];                      // [0]=sum(qhat), [1]=sum e_k*Vb
  float* frow = full + (size_t)n * Gg * Cc;
  const float* clsr = xin + (size_t)n * Gg * Cc + h * Dd;

  if (tid < Dd) {
    float a = 0.f;
    for (int d = 0; d < Dd; ++d) a += clsr[d] * Qw[(size_t)(tid * HEADS + h) * Dd + d];
    qhat[tid] = a + Qb[h];
  }
  __syncthreads();
  if (tid < Dd) {
    float a = 0.f;
    for (int xx = 0; xx < Dd; ++xx) a += qhat[xx] * Kw[(size_t)(xx * HEADS + h) * Dd + tid];
    wtil[tid] = a;
  }
  if (tid == 0) {
    float q = 0.f;
    for (int xx = 0; xx < Dd; ++xx) q += qhat[xx];
    scal[0] = q;
  }
  __syncthreads();

  float s = -3.402823466e38f;
  if (tid < Gg) {
    const float* kr = frow + (size_t)tid * Cc + h * Dd;
    float a = 0.f;
    for (int d = 0; d < Dd; ++d) a += kr[d] * wtil[d];
    s = (a + Kb[tid * HEADS + h] * scal[0]) * 0.125f;   // 1/sqrt(64)
  }
  red[tid] = s;
  __syncthreads();
  for (int off = 128; off > 0; off >>= 1) {
    if (tid < off) red[tid] = fmaxf(red[tid], red[tid + off]);
    __syncthreads();
  }
  const float mx = red[0];
  __syncthreads();
  const float e = (tid < Gg) ? expf(s - mx) : 0.f;
  sc[tid] = e; red[tid] = e;
  __syncthreads();
  for (int off = 128; off > 0; off >>= 1) {
    if (tid < off) red[tid] += red[tid + off];
    __syncthreads();
  }
  const float invs = 1.f / red[0];

  if (tid < Dd) {
    float u = 0.f;
    for (int k = 0; k < Gg; ++k) u += sc[k] * frow[(size_t)k * Cc + h * Dd + tid];
    wtil[tid] = u;                                  // reuse for u[d]
  }
  if (tid == 0) {
    float vb = 0.f;
    for (int k = 0; k < Gg; ++k) vb += sc[k] * Vb[k * HEADS + h];
    scal[1] = vb;
  }
  __syncthreads();
  if (tid < Dd) {
    float o = scal[1];
    for (int d = 0; d < Dd; ++d) o += wtil[d] * Vw[(size_t)(tid * HEADS + h) * Dd + d];
    frow[h * Dd + tid] = o * invs;                  // new_cls in place
  }
}

// ---------------------------------------------------------------------------
// K5: xo = full + x (virtual); LayerNorm -> f16 activations for the MLP.
// One block per row (768 = 3*256).
// ---------------------------------------------------------------------------
__global__ __launch_bounds__(256) void ln_kernel(const float* __restrict__ full,
                                                 const float* __restrict__ xin,
                                                 const float* __restrict__ w,
                                                 const float* __restrict__ b,
                                                 _Float16* __restrict__ yh) {
  __shared__ float r1[256], r2[256];
  const size_t base = (size_t)blockIdx.x * Cc;
  const int tid = threadIdx.x;
  const float f0 = full[base + tid]       + xin[base + tid];
  const float f1 = full[base + tid + 256] + xin[base + tid + 256];
  const float f2 = full[base + tid + 512] + xin[base + tid + 512];
  r1[tid] = f0 + f1 + f2;
  r2[tid] = f0 * f0 + f1 * f1 + f2 * f2;
  __syncthreads();
  for (int off = 128; off > 0; off >>= 1) {
    if (tid < off) { r1[tid] += r1[tid + off]; r2[tid] += r2[tid + off]; }
    __syncthreads();
  }
  const float mu   = r1[0] * (1.f / Cc);
  const float var  = r2[0] * (1.f / Cc) - mu * mu;
  const float rstd = rsqrtf(var + 1e-6f);
  yh[base + tid]       = (_Float16)((f0 - mu) * rstd * w[tid]       + b[tid]);
  yh[base + tid + 256] = (_Float16)((f1 - mu) * rstd * w[tid + 256] + b[tid + 256]);
  yh[base + tid + 512] = (_Float16)((f2 - mu) * rstd * w[tid + 512] + b[tid + 512]);
}

// ---------------------------------------------------------------------------
// WMMA tile loaders (wave32, contiguous k order shared by A and B):
//  A 16x32 f16: lane m = lane&15, half = lane>>4; VGPR j -> K = 2j + 16*half
//               => two adjacent 16B chunks of row m.
//  B 32x16 f16: lane n = lane&15; VGPR j -> K = 2j + 16*half => one 32B chunk
//               from the N-major (transposed) weight array.
// ---------------------------------------------------------------------------
DEVFN v16h load_a_16x32(const _Float16* __restrict__ A, int ldk, int m, int k0, int hl) {
  const _Float16* p = A + (size_t)m * ldk + k0 + 16 * hl;
  v8h lo = *(const v8h*)p;            // K = 16h .. 16h+7
  v8h hi = *(const v8h*)(p + 8);      // K = 16h+8 .. 16h+15
  return __builtin_shufflevector(lo, hi, 0,1,2,3,4,5,6,7,8,9,10,11,12,13,14,15);
}
DEVFN v16h load_b_32x16(const _Float16* __restrict__ Bt, int ldk, int nrow, int k0, int hl) {
  return *(const v16h*)(Bt + (size_t)nrow * ldk + k0 + 16 * hl);
}

// K6: h1 = gelu_exact(y @ w1 + b1), f16 out. One wave -> 16x64 tile.
__global__ __launch_bounds__(32) void gemm_mlp1(const _Float16* __restrict__ A,
                                                const _Float16* __restrict__ Bt,
                                                const float* __restrict__ bias,
                                                _Float16* __restrict__ H) {
  const int m0 = blockIdx.x * 16, n0 = blockIdx.y * 64;
  const int ln = threadIdx.x & 31;
  const int hl = ln >> 4, q = ln & 15;
  v8f c0 = {}, c1 = {}, c2 = {}, c3 = {};
  for (int k0 = 0; k0 < Cc; k0 += 32) {
    __builtin_prefetch(A + (size_t)(m0 + q) * Cc + k0 + 64, 0, 1);
    v16h a  = load_a_16x32(A, Cc, m0 + q, k0, hl);
    v16h b0 = load_b_32x16(Bt, Cc, n0 + q,      k0, hl);
    v16h b1 = load_b_32x16(Bt, Cc, n0 + 16 + q, k0, hl);
    v16h b2 = load_b_32x16(Bt, Cc, n0 + 32 + q, k0, hl);
    v16h b3 = load_b_32x16(Bt, Cc, n0 + 48 + q, k0, hl);
    c0 = __builtin_amdgcn_wmma_f32_16x16x32_f16(false, a, false, b0, (short)0, c0, false, false);
    c1 = __builtin_amdgcn_wmma_f32_16x16x32_f16(false, a, false, b1, (short)0, c1, false, false);
    c2 = __builtin_amdgcn_wmma_f32_16x16x32_f16(false, a, false, b2, (short)0, c2, false, false);
    c3 = __builtin_amdgcn_wmma_f32_16x16x32_f16(false, a, false, b3, (short)0, c3, false, false);
  }
#pragma unroll
  for (int t = 0; t < 4; ++t) {
    v8f cc = (t == 0) ? c0 : (t == 1) ? c1 : (t == 2) ? c2 : c3;
    const int nn = n0 + t * 16 + q;
    const float bn = bias[nn];
#pragma unroll
    for (int r = 0; r < 8; ++r) {
      const int mm = m0 + r + 8 * hl;
      const float v = cc[r] + bn;
      const float g = 0.5f * v * (1.f + erff(v * 0.70710678118654752f));
      H[(size_t)mm * MLPd + nn] = (_Float16)g;
    }
  }
}

// K7: out = (full + x) + h1 @ w2 + b2  (final residual folded into epilogue).
__global__ __launch_bounds__(32) void gemm_mlp2(const _Float16* __restrict__ A,
                                                const _Float16* __restrict__ Bt,
                                                const float* __restrict__ bias,
                                                const float* __restrict__ full,
                                                const float* __restrict__ xin,
                                                float* __restrict__ out) {
  const int m0 = blockIdx.x * 16, n0 = blockIdx.y * 64;
  const int ln = threadIdx.x & 31;
  const int hl = ln >> 4, q = ln & 15;
  v8f c0 = {}, c1 = {}, c2 = {}, c3 = {};
  for (int k0 = 0; k0 < MLPd; k0 += 32) {
    __builtin_prefetch(A + (size_t)(m0 + q) * MLPd + k0 + 64, 0, 1);
    v16h a  = load_a_16x32(A, MLPd, m0 + q, k0, hl);
    v16h b0 = load_b_32x16(Bt, MLPd, n0 + q,      k0, hl);
    v16h b1 = load_b_32x16(Bt, MLPd, n0 + 16 + q, k0, hl);
    v16h b2 = load_b_32x16(Bt, MLPd, n0 + 32 + q, k0, hl);
    v16h b3 = load_b_32x16(Bt, MLPd, n0 + 48 + q, k0, hl);
    c0 = __builtin_amdgcn_wmma_f32_16x16x32_f16(false, a, false, b0, (short)0, c0, false, false);
    c1 = __builtin_amdgcn_wmma_f32_16x16x32_f16(false, a, false, b1, (short)0, c1, false, false);
    c2 = __builtin_amdgcn_wmma_f32_16x16x32_f16(false, a, false, b2, (short)0, c2, false, false);
    c3 = __builtin_amdgcn_wmma_f32_16x16x32_f16(false, a, false, b3, (short)0, c3, false, false);
  }
#pragma unroll
  for (int t = 0; t < 4; ++t) {
    v8f cc = (t == 0) ? c0 : (t == 1) ? c1 : (t == 2) ? c2 : c3;
    const int nn = n0 + t * 16 + q;
    const float bn = bias[nn];
#pragma unroll
    for (int r = 0; r < 8; ++r) {
      const int mm = m0 + r + 8 * hl;
      const size_t idx = (size_t)mm * Cc + nn;
      out[idx] = full[idx] + xin[idx] + cc[r] + bn;
    }
  }
}

// ---------------------------------------------------------------------------
extern "C" void kernel_launch(void* const* d_in, const int* in_sizes, int n_in,
                              void* d_out, int out_size, void* d_ws, size_t ws_size,
                              hipStream_t stream) {
  (void)in_sizes; (void)n_in; (void)out_size; (void)ws_size;
  const float* x     = (const float*)d_in[0];
  const float* mixer = (const float*)d_in[1];
  const float* Qw    = (const float*)d_in[2];
  const float* Qb    = (const float*)d_in[3];
  const float* Kw    = (const float*)d_in[4];
  const float* Kb    = (const float*)d_in[5];
  const float* Vw    = (const float*)d_in[6];
  const float* Vb    = (const float*)d_in[7];
  const float* ln_w  = (const float*)d_in[8];
  const float* ln_b  = (const float*)d_in[9];
  const float* w1    = (const float*)d_in[10];
  const float* b1    = (const float*)d_in[11];
  const float* w2    = (const float*)d_in[12];
  const float* b2    = (const float*)d_in[13];
  float* out = (float*)d_out;

  char* ws = (char*)d_ws;
  size_t off = 0;
  auto alloc = [&](size_t bytes) -> void* {
    void* p = ws + off;
    off = (off + bytes + 255) & ~(size_t)255;
    return p;
  };
  float*    full = (float*)   alloc((size_t)Mrows * Cc * sizeof(float));       // 19.4 MB
  float*    meff = (float*)   alloc((size_t)MIXO * 2 * sizeof(float));         // 0.7 MB
  _Float16* w1t  = (_Float16*)alloc((size_t)Cc * MLPd * sizeof(_Float16));     // 4.7 MB
  _Float16* w2t  = (_Float16*)alloc((size_t)MLPd * Cc * sizeof(_Float16));     // 4.7 MB
  _Float16* yh   = (_Float16*)alloc((size_t)Mrows * Cc * sizeof(_Float16));    // 9.7 MB
  _Float16* h1h  = (_Float16*)alloc((size_t)Mrows * MLPd * sizeof(_Float16));  // 38.7 MB

  // 1) mixer -> m_eff (dominant HBM traffic: 528 MB)
  mixsum_kernel<<<dim3((MIXO + 7) / 8), dim3(256), 0, stream>>>(mixer, meff);
  // 2) weights -> transposed f16 (independent of 1)
  transpose_f16_kernel<<<dim3(Cc / 32, MLPd / 32), dim3(32, 8), 0, stream>>>(w1, w1t, Cc, MLPd);
  transpose_f16_kernel<<<dim3(MLPd / 32, Cc / 32), dim3(32, 8), 0, stream>>>(w2, w2t, MLPd, Cc);
  // 3) spectral mixing -> full (rows 1..196) + cls copy (row 0)
  spectral_kernel<<<dim3(Nn, Cc / 64), dim3(64), 0, stream>>>(x, meff, full);
  // 4) single-query attention, new_cls written in place into full row 0
  attn_kernel<<<dim3(Nn, HEADS), dim3(256), 0, stream>>>(x, Qw, Qb, Kw, Kb, Vw, Vb, full);
  // 5) layernorm(full + x) -> f16 activations
  ln_kernel<<<dim3(Mrows), dim3(256), 0, stream>>>(full, x, ln_w, ln_b, yh);
  // 6) MLP GEMM1 (+bias, exact gelu) via WMMA f16
  gemm_mlp1<<<dim3(Mrows / 16, MLPd / 64), dim3(32), 0, stream>>>(yh, w1t, b1, h1h);
  // 7) MLP GEMM2 (+bias, + residual full + x) via WMMA f16 -> d_out
  gemm_mlp2<<<dim3(Mrows / 16, Cc / 64), dim3(32), 0, stream>>>(h1h, w2t, b2, full, x, out);
}
